// RSSM_31645319037114
// MI455X (gfx1250) — matrix operand order repaired and checked by reading, not compiled, not run on hardware
//
#include <hip/hip_runtime.h>
#include <math.h>

// ---------------- problem constants (fixed by the reference) ----------------
#define H_DIM 2048
#define S_DIM 1024
#define A_DIM 2048
#define X_DIM 3072   // A_DIM + S_DIM
#define G_DIM 6144   // 3*H
#define BR    256
#define DEPTH 16
#define OUT_STRIDE 2051

// ---------------- CDNA5 WMMA types ----------------
typedef __attribute__((ext_vector_type(16))) __bf16 v16bf;
typedef __attribute__((ext_vector_type(8)))  float  v8f;
typedef __attribute__((ext_vector_type(4)))  unsigned int u32x4;

union FragBF {
  v16bf v;
  u32x4 u[2];
};

__device__ __forceinline__ unsigned short f2bf(float f) {
  unsigned int u = __float_as_uint(f);
  u += 0x7FFFu + ((u >> 16) & 1u);     // round-to-nearest-even
  return (unsigned short)(u >> 16);
}
__device__ __forceinline__ float bf2f(unsigned short s) {
  return __uint_as_float(((unsigned int)s) << 16);
}

// ---------------------------------------------------------------------------
// GEMM: C[M=256, N] = act( A[M,K](bf16, K-contig) * W[N,K]^T (bf16) + bias )
//
// Block = 256 threads = 8 waves arranged 4(M) x 2(N); block tile 128x128.
// Each wave owns a 32x64 strip: 2 M-tiles x 4 N-tiles = 8 fp32 accumulators,
// 8 x v_wmma_f32_16x16x32_bf16 per 32-wide K step.
//
// K loop is unrolled x2 with register ping-pong double buffering: next
// chunk's 12 b128 loads are issued before the current chunk's WMMAs, so the
// hardware waits become partial (s_wait_loadcnt 12) and L2 weight streaming
// overlaps the matrix pipe.  Weight rows (out,in layout) are K-contiguous,
// matching the WMMA B per-lane layout -> no transpose loads needed.
//
// 128x128 block tiles: weights (the ~88MB L2-resident operand) are re-read
// only 2x per step instead of 8x with skinnier M tiles.
//
// MODE 0: fp32 out; 1: bf16 out; 2: bf16 out with exact GELU fused.
// ---------------------------------------------------------------------------
template <int MODE>
__global__ __launch_bounds__(256) void gemm_wmma_bf16(
    const unsigned short* __restrict__ A,
    const unsigned short* __restrict__ W,
    const float* __restrict__ bias,
    void* __restrict__ Cout, int K) {
  const int lane   = threadIdx.x & 31;
  const int wid    = threadIdx.x >> 5;
  const int mBase  = blockIdx.y * 128 + (wid & 3) * 32;   // 4 waves in M
  const int nBase  = blockIdx.x * 128 + (wid >> 2) * 64;  // 2 waves in N
  const int l15    = lane & 15;
  const int hiHalf = lane >> 4;
  const int N      = gridDim.x * 128;

  // Per-lane fragment base pointers (ISA 16-bit A / B wave32 layouts).
  const unsigned short* Arow[2];
  const unsigned short* Brow[4];
#pragma unroll
  for (int mt = 0; mt < 2; ++mt)
    Arow[mt] = A + (size_t)(mBase + mt * 16 + l15) * K + hiHalf * 8;
#pragma unroll
  for (int nt = 0; nt < 4; ++nt)
    Brow[nt] = W + (size_t)(nBase + nt * 16 + l15) * K + hiHalf * 8;

  const v8f vzero = {0.f, 0.f, 0.f, 0.f, 0.f, 0.f, 0.f, 0.f};
  v8f acc[2][4];
#pragma unroll
  for (int mt = 0; mt < 2; ++mt)
#pragma unroll
    for (int nt = 0; nt < 4; ++nt) acc[mt][nt] = vzero;

  auto loadA = [&](FragBF (&fa)[2], int k) {
#pragma unroll
    for (int mt = 0; mt < 2; ++mt) {
      fa[mt].u[0] = *(const u32x4*)(Arow[mt] + k);       // K = k+{0..7} / {8..15}
      fa[mt].u[1] = *(const u32x4*)(Arow[mt] + k + 16);  // K = k+{16..23}/{24..31}
    }
  };
  auto loadB = [&](FragBF (&fb)[4], int k) {
#pragma unroll
    for (int nt = 0; nt < 4; ++nt) {
      const unsigned short* p = Brow[nt] + k;
      fb[nt].u[0] = *(const u32x4*)(p);
      fb[nt].u[1] = *(const u32x4*)(p + 16);
    }
  };
  auto mma = [&](FragBF (&fa)[2], FragBF (&fb)[4]) {
#pragma unroll
    for (int mt = 0; mt < 2; ++mt)
#pragma unroll
      for (int nt = 0; nt < 4; ++nt)
        acc[mt][nt] = __builtin_amdgcn_wmma_f32_16x16x32_bf16(
            false, fa[mt].v, false, fb[nt].v, (short)0, acc[mt][nt], false, false);
  };

  FragBF aP[2], bP[4], aQ[2], bQ[4];
  loadA(aP, 0);
  loadB(bP, 0);
  // K is always a multiple of 64 here (2048 / 3072).
  for (int k0 = 0; k0 < K; k0 += 64) {
    loadA(aQ, k0 + 32);
    loadB(bQ, k0 + 32);
    if (k0 + 128 < K) {  // prefetch the streaming (weight) operand two chunks ahead
#pragma unroll
      for (int nt = 0; nt < 4; ++nt) __builtin_prefetch(Brow[nt] + k0 + 128, 0, 3);
    }
    mma(aP, bP);                       // waits only on aP/bP; aQ/bQ still in flight
    const int kn = k0 + 64;
    if (kn < K) {
      loadA(aP, kn);
      loadB(bP, kn);
    }
    mma(aQ, bQ);                       // waits only on aQ/bQ; aP/bP in flight
  }

  // C/D layout: VGPR i -> M = base + i (+8 for upper lane half), N = base + l15
  const int mOff = hiHalf * 8;
#pragma unroll
  for (int mt = 0; mt < 2; ++mt) {
    const int mRow = mBase + mt * 16 + mOff;
#pragma unroll
    for (int nt = 0; nt < 4; ++nt) {
      const int n = nBase + nt * 16 + l15;
      const float bv = bias[n];
#pragma unroll
      for (int i = 0; i < 8; ++i) {
        float v = acc[mt][nt][i] + bv;
        const size_t off = (size_t)(mRow + i) * N + n;
        if (MODE == 0) {
          ((float*)Cout)[off] = v;
        } else if (MODE == 1) {
          ((unsigned short*)Cout)[off] = f2bf(v);
        } else {
          float g = 0.5f * v * (1.0f + erff(v * 0.70710678118654752f));  // exact GELU
          ((unsigned short*)Cout)[off] = f2bf(g);
        }
      }
    }
  }
}

// ---------------- fp32 -> bf16 weight conversion ----------------
__global__ __launch_bounds__(256) void cvt_f32_bf16(const float* __restrict__ s,
                                                    unsigned short* __restrict__ d,
                                                    int n) {
  int i = blockIdx.x * 256 + threadIdx.x;
  const int stride = gridDim.x * 256;
  for (; i < n; i += stride) d[i] = f2bf(s[i]);
}

// ---------------- initial state broadcast ----------------
__global__ __launch_bounds__(256) void init_state(
    const float* __restrict__ action, const float* __restrict__ h0,
    const float* __restrict__ s0, unsigned short* __restrict__ xb,
    unsigned short* __restrict__ hbf) {
  const int idx = blockIdx.x * 256 + threadIdx.x;
  if (idx < BR * X_DIM) {
    const int j = idx % X_DIM;
    xb[idx] = f2bf(j < A_DIM ? action[j] : s0[j - A_DIM]);
  }
  if (idx < BR * H_DIM) {
    hbf[idx] = f2bf(h0[idx & (H_DIM - 1)]);
  }
}

// ---------------- GRU gate combine ----------------
__global__ __launch_bounds__(256) void gru_combine(
    const float* __restrict__ gi, const float* __restrict__ gh,
    unsigned short* __restrict__ hbf, unsigned short* __restrict__ hs) {
  const int idx = blockIdx.x * 256 + threadIdx.x;
  if (idx >= BR * H_DIM) return;
  const int b = idx >> 11, j = idx & (H_DIM - 1);
  const float* gib = gi + (size_t)b * G_DIM;
  const float* ghb = gh + (size_t)b * G_DIM;
  const float r = 1.0f / (1.0f + expf(-(gib[j] + ghb[j])));
  const float z = 1.0f / (1.0f + expf(-(gib[H_DIM + j] + ghb[H_DIM + j])));
  const float n = tanhf(gib[2 * H_DIM + j] + r * ghb[2 * H_DIM + j]);
  const float hOld = bf2f(hbf[idx]);
  const float hN = (1.0f - z) * n + z * hOld;
  const unsigned short hb = f2bf(hN);
  hbf[idx] = hb;
  hs[(size_t)b * X_DIM + j] = hb;
}

// ---------------- per-categorical softmax (32 classes) ----------------
__global__ __launch_bounds__(256) void softmax32(
    const float* __restrict__ prior, unsigned short* __restrict__ hs,
    unsigned short* __restrict__ xb) {
  const int idx = blockIdx.x * 256 + threadIdx.x;  // BR*32 = 8192
  if (idx >= BR * 32) return;
  const int b = idx >> 5, cat = idx & 31;
  const float* p = prior + (size_t)b * S_DIM + cat * 32;
  float mx = p[0];
#pragma unroll
  for (int c = 1; c < 32; ++c) mx = fmaxf(mx, p[c]);
  float e[32];
  float sum = 0.0f;
#pragma unroll
  for (int c = 0; c < 32; ++c) { e[c] = expf(p[c] - mx); sum += e[c]; }
  const float inv = 1.0f / sum;
  const size_t base = (size_t)b * X_DIM + H_DIM + (size_t)cat * 32;
#pragma unroll
  for (int c = 0; c < 32; ++c) {
    const unsigned short v = f2bf(e[c] * inv);
    hs[base + c] = v;
    xb[base + c] = v;
  }
}

// ---------------- heads (r/v/t dot products via LDS reduce) + output ----------------
__global__ __launch_bounds__(256) void heads_out(
    const unsigned short* __restrict__ hs, const float* __restrict__ ns,
    const float* __restrict__ W_r, const float* __restrict__ b_r,
    const float* __restrict__ W_v, const float* __restrict__ b_v,
    const float* __restrict__ W_t, const float* __restrict__ b_t,
    float* __restrict__ out, unsigned short* __restrict__ xb, int step) {
  __shared__ float sr[256], sv[256], st[256];
  const int b = blockIdx.x;
  const int tid = threadIdx.x;
  const unsigned short* hsb = hs + (size_t)b * X_DIM;
  float pr = 0.f, pv = 0.f, pt = 0.f;
  for (int j = tid; j < X_DIM; j += 256) {
    const float x = bf2f(hsb[j]);
    pr += x * W_r[j];
    pv += x * W_v[j];
    pt += x * W_t[j];
  }
  sr[tid] = pr; sv[tid] = pv; st[tid] = pt;
  __syncthreads();
  for (int off = 128; off > 0; off >>= 1) {
    if (tid < off) {
      sr[tid] += sr[tid + off];
      sv[tid] += sv[tid + off];
      st[tid] += st[tid + off];
    }
    __syncthreads();
  }
  float* ob = out + ((size_t)b * DEPTH + step) * OUT_STRIDE;
  if (tid == 0) {
    ob[0] = tanhf(sr[0] + b_r[0]);
    ob[1] = sv[0] + b_v[0];
    ob[2] = 1.0f / (1.0f + expf(-(st[0] + b_t[0])));
  }
  const float* nsb = ns + (size_t)b * A_DIM;
  unsigned short* xbb = xb + (size_t)b * X_DIM;
  for (int j = tid; j < A_DIM; j += 256) {
    const float x = nsb[j];
    ob[3 + j] = x;           // output next_state_embd
    xbb[j] = f2bf(x);        // next action = ns
  }
}

// ---------------- host-side GEMM dispatch ----------------
static inline void launch_gemm(int mode, const unsigned short* A, const unsigned short* W,
                               const float* bias, void* C, int N, int K, hipStream_t s) {
  dim3 grid(N / 128, BR / 128);
  switch (mode) {
    case 0: gemm_wmma_bf16<0><<<grid, 256, 0, s>>>(A, W, bias, C, K); break;
    case 1: gemm_wmma_bf16<1><<<grid, 256, 0, s>>>(A, W, bias, C, K); break;
    default: gemm_wmma_bf16<2><<<grid, 256, 0, s>>>(A, W, bias, C, K); break;
  }
}

extern "C" void kernel_launch(void* const* d_in, const int* in_sizes, int n_in,
                              void* d_out, int out_size, void* d_ws, size_t ws_size,
                              hipStream_t stream) {
  (void)in_sizes; (void)n_in; (void)out_size; (void)ws_size;
  const float* action = (const float*)d_in[0];
  const float* h0     = (const float*)d_in[1];
  const float* s0     = (const float*)d_in[2];
  const float* W_ap   = (const float*)d_in[3];
  const float* b_ap   = (const float*)d_in[4];
  const float* W_ih   = (const float*)d_in[5];
  const float* b_ih   = (const float*)d_in[6];
  const float* W_hh   = (const float*)d_in[7];
  const float* b_hh   = (const float*)d_in[8];
  const float* W_p1   = (const float*)d_in[9];
  const float* b_p1   = (const float*)d_in[10];
  const float* W_p2   = (const float*)d_in[11];
  const float* b_p2   = (const float*)d_in[12];
  const float* W_r    = (const float*)d_in[13];
  const float* b_r    = (const float*)d_in[14];
  const float* W_v    = (const float*)d_in[15];
  const float* b_v    = (const float*)d_in[16];
  const float* W_t    = (const float*)d_in[17];
  const float* b_t    = (const float*)d_in[18];
  const float* W_ns   = (const float*)d_in[19];
  const float* b_ns   = (const float*)d_in[20];
  float* out = (float*)d_out;

  // ---- workspace carve-up (256B aligned) ----
  char* w = (char*)d_ws;
  auto take = [&](size_t bytes) -> void* {
    char* p = w;
    w += (bytes + 255) & ~(size_t)255;
    return p;
  };
  unsigned short* wap = (unsigned short*)take((size_t)H_DIM * X_DIM * 2);
  unsigned short* wih = (unsigned short*)take((size_t)G_DIM * H_DIM * 2);
  unsigned short* whh = (unsigned short*)take((size_t)G_DIM * H_DIM * 2);
  unsigned short* wp1 = (unsigned short*)take((size_t)H_DIM * H_DIM * 2);
  unsigned short* wp2 = (unsigned short*)take((size_t)S_DIM * H_DIM * 2);
  unsigned short* wns = (unsigned short*)take((size_t)A_DIM * X_DIM * 2);
  unsigned short* xb  = (unsigned short*)take((size_t)BR * X_DIM * 2);  // [a | s]
  unsigned short* hbf = (unsigned short*)take((size_t)BR * H_DIM * 2);
  unsigned short* inp = (unsigned short*)take((size_t)BR * H_DIM * 2);
  float*          gi  = (float*)take((size_t)BR * G_DIM * 4);
  float*          gh  = (float*)take((size_t)BR * G_DIM * 4);
  unsigned short* p1  = (unsigned short*)take((size_t)BR * H_DIM * 2);
  float*          pri = (float*)take((size_t)BR * S_DIM * 4);
  unsigned short* hs  = (unsigned short*)take((size_t)BR * X_DIM * 2);  // [h_new | s_new]
  float*          ns  = (float*)take((size_t)BR * A_DIM * 4);

  // ---- convert weights fp32 -> bf16 (L2-resident ~88MB working set) ----
  auto cvt = [&](const float* src, unsigned short* dst, int n) {
    int blocks = (n + 255) / 256;
    if (blocks > 4096) blocks = 4096;
    cvt_f32_bf16<<<blocks, 256, 0, stream>>>(src, dst, n);
  };
  cvt(W_ap, wap, H_DIM * X_DIM);
  cvt(W_ih, wih, G_DIM * H_DIM);
  cvt(W_hh, whh, G_DIM * H_DIM);
  cvt(W_p1, wp1, H_DIM * H_DIM);
  cvt(W_p2, wp2, S_DIM * H_DIM);
  cvt(W_ns, wns, A_DIM * X_DIM);

  init_state<<<(BR * X_DIM + 255) / 256, 256, 0, stream>>>(action, h0, s0, xb, hbf);

  // ---- 16 sequential rollout steps ----
  for (int step = 0; step < DEPTH; ++step) {
    // inp = Linear([a,s]; W_ap)                (bf16 out)
    launch_gemm(1, xb, wap, b_ap, inp, H_DIM, X_DIM, stream);
    // gi = Linear(inp; W_ih), gh = Linear(h; W_hh)   (fp32 out)
    launch_gemm(0, inp, wih, b_ih, gi, G_DIM, H_DIM, stream);
    launch_gemm(0, hbf, whh, b_hh, gh, G_DIM, H_DIM, stream);
    // GRU combine -> h_new (updates hbf, hs[:,0:2048])
    gru_combine<<<(BR * H_DIM + 255) / 256, 256, 0, stream>>>(gi, gh, hbf, hs);
    // p1 = GELU(Linear(h_new; W_p1))           (bf16 out, fused exact GELU)
    launch_gemm(2, hbf, wp1, b_p1, p1, H_DIM, H_DIM, stream);
    // prior = Linear(p1; W_p2)                 (fp32 out)
    launch_gemm(0, p1, wp2, b_p2, pri, S_DIM, H_DIM, stream);
    // s_new = softmax32(prior) -> hs[:,2048:], xb[:,2048:]
    softmax32<<<(BR * 32 + 255) / 256, 256, 0, stream>>>(pri, hs, xb);
    // ns = Linear(hs; W_ns)                    (fp32 out)
    launch_gemm(0, hs, wns, b_ns, ns, A_DIM, X_DIM, stream);
    // heads + write outputs + next action = ns
    heads_out<<<BR, 256, 0, stream>>>(hs, ns, W_r, b_r, W_v, b_v, W_t, b_t,
                                      out, xb, step);
  }
}
